// DGCNNModel_66503273611675
// MI455X (gfx1250) — matrix operand order repaired and checked by reading, not compiled, not run on hardware
//
#include <hip/hip_runtime.h>
#include <math.h>

// ---------------------------------------------------------------------------
// DGCNN forward on gfx1250 (MI455X).
//  - bf16 WMMA (v_wmma_f32_16x16x32_bf16) for all dense GEMMs, K-loop fully
//    unrolled via templates (4 / 1 / 11 static WMMAs per instantiation).
//  - A stripes staged to LDS per workgroup: TDM tensor_load_to_lds when the
//    builtin exists, coalesced global->LDS copy otherwise.
//  - Edge scatter with global_atomic_add_f32 (L2-resident), wave-per-graph
//    stable top-K sort-pool, scalar CNN/MLP head.
// ---------------------------------------------------------------------------

typedef __attribute__((ext_vector_type(16))) __bf16 bf16x16;
typedef __attribute__((ext_vector_type(8)))  float  f32x8;
typedef unsigned int u32x4 __attribute__((ext_vector_type(4)));
typedef int          i32x8 __attribute__((ext_vector_type(8)));
typedef int          i32x4 __attribute__((ext_vector_type(4)));

#define NGRAPH 2048
#define TOPK   30
#define FEAT   97   // 32 + 32 + 32 + 1

#if defined(__has_builtin)
#if __has_builtin(__builtin_amdgcn_tensor_load_to_lds) && __has_builtin(__builtin_amdgcn_s_wait_tensorcnt)
#define HAVE_TDM 1
#endif
#endif
#ifndef HAVE_TDM
#define HAVE_TDM 0
#endif

static inline int cdiv(int a, int b) { return (a + b - 1) / b; }

#if HAVE_TDM
// 1-D contiguous tile: nelem f32 from gptr -> LDS at lptr. Descriptor per
// CDNA5 ISA (count=1, type=2, data_size=4B).
__device__ __forceinline__ void tdm_copy_1d(const float* gptr, float* lptr, int nelem) {
  unsigned long long ga = (unsigned long long)(uintptr_t)gptr;
  unsigned int lds = (unsigned int)(uintptr_t)lptr;   // low 32 bits = LDS offset

  u32x4 g0;
  g0[0] = 1u;                                         // count=1, user mode
  g0[1] = lds;                                        // lds_addr
  g0[2] = (unsigned int)(ga & 0xFFFFFFFFull);         // global_addr[31:0]
  g0[3] = (unsigned int)((ga >> 32) & 0x1FFFFFFull) | (2u << 30);  // addr[56:32] | type=2

  unsigned long long dim0 = (unsigned long long)(unsigned)nelem;
  unsigned long long tile0 = (unsigned long long)(unsigned)nelem;  // <= 8192, fits 16b
  unsigned long long str0 = dim0;
  unsigned long long q0 = (2ull << 16)                     // data_size = 4B
                        | ((dim0 & 0xFFFFull) << 48);      // tensor_dim0[15:0]
  unsigned long long q1 = ((dim0 >> 16) & 0xFFFFull)       // tensor_dim0[31:16]
                        | (1ull << 16)                     // tensor_dim1 = 1
                        | (tile0 << 48);                   // tile_dim0
  unsigned long long q2 = (str0 & 0xFFFFFFFFull) << 32;    // tile_dim1/2 = 0, stride0 lo
  unsigned long long q3 = (str0 >> 32) & 0xFFFFull;        // stride0 hi, stride1 = 0

  i32x8 g1;
  g1[0] = (int)(q0 & 0xFFFFFFFFull); g1[1] = (int)(q0 >> 32);
  g1[2] = (int)(q1 & 0xFFFFFFFFull); g1[3] = (int)(q1 >> 32);
  g1[4] = (int)(q2 & 0xFFFFFFFFull); g1[5] = (int)(q2 >> 32);
  g1[6] = (int)(q3 & 0xFFFFFFFFull); g1[7] = (int)(q3 >> 32);

  i32x4 gz = {0, 0, 0, 0};
#if __clang_major__ >= 23
  i32x8 gz8 = {0, 0, 0, 0, 0, 0, 0, 0};
  __builtin_amdgcn_tensor_load_to_lds(g0, g1, gz, gz, gz8, 0);
#else
  __builtin_amdgcn_tensor_load_to_lds(g0, g1, gz, gz, 0);
#endif
  __builtin_amdgcn_s_wait_tensorcnt(0);
}
#endif

// -------------------------- degree / dinv ----------------------------------
__global__ void fill_f32_k(float* p, float v, int n) {
  int i = blockIdx.x * blockDim.x + threadIdx.x;
  if (i < n) p[i] = v;
}

__global__ void deg_accum_k(const int* __restrict__ dst, float* __restrict__ deg, int e) {
  int i = blockIdx.x * blockDim.x + threadIdx.x;
  if (i < e) atomicAdd(&deg[dst[i]], 1.0f);
}

__global__ void deg_to_dinv_k(float* d, int n) {
  int i = blockIdx.x * blockDim.x + threadIdx.x;
  if (i < n) d[i] = rsqrtf(d[i]);
}

// -------------------------- bf16 WMMA GEMM ---------------------------------
// D(M,N) = act(A(M,K) @ B(K,N) [+ bias]).  One 16x16 D tile per wave32.
// Block = 8 waves covering a ROWS x N output stripe; A stripe staged in LDS
// (TDM when available), B staged in LDS when small (<=16KB).
// Register layouts used here (A/B k order is this code's choice, not the ISA table's):
//   A 16x32: lane = row m (both halves); half selects K 0-15
//            vs 16-31, elem j -> K = j + 16*half.
//   B 32x16: lanes 0..15 hold K=0..15, lanes 16..31 K=16..31; col = lane&15.
//   C/D:     vgpr r, lane l -> row = r + 8*(l>>4), col = l&15.
template <int K, int N, int ACT, bool BIAS>
__global__ __launch_bounds__(256) void gemm_bf16_wmma_k(const float* __restrict__ A,
                                                        const float* __restrict__ Bm,
                                                        const float* __restrict__ bias,
                                                        float* __restrict__ D) {
  constexpr int TILES_N = N / 16;
  constexpr int ROWS    = (8 / TILES_N) * 16;          // rows per block
  constexpr bool STAGE_B = (K * N * 4 <= 16384);
  constexpr int LBSZ = STAGE_B ? K * N : 16;

  __shared__ float lA[ROWS * K];
  __shared__ float lB[LBSZ];

  const int tid = threadIdx.x;
  const int w = tid >> 5;
  const size_t base = (size_t)blockIdx.x * ROWS * K;

  // ---- stage A stripe (contiguous ROWS*K floats) ----
#if HAVE_TDM
  if (w == 0) tdm_copy_1d(A + base, lA, ROWS * K);
#else
  for (int idx = tid; idx < ROWS * K; idx += 256) lA[idx] = A[base + idx];
#endif
  if constexpr (STAGE_B) {
    for (int idx = tid; idx < K * N; idx += 256) lB[idx] = Bm[idx];
  }
  __syncthreads();

  // ---- compute one 16x16 tile per wave ----
  const int lane = tid & 31;
  const int half = lane >> 4;
  const int m    = lane & 15;
  const int rowTile = w / TILES_N;
  const int colTile = w % TILES_N;
  const int c0 = colTile * 16;

  f32x8 acc = {0.f, 0.f, 0.f, 0.f, 0.f, 0.f, 0.f, 0.f};
#pragma unroll
  for (int k0 = 0; k0 < K; k0 += 32) {
    bf16x16 a, b;
#pragma unroll
    for (int j = 0; j < 16; ++j) {
      const int ka = k0 + j + 16 * half;
      a[j] = (__bf16)lA[(rowTile * 16 + m) * K + ka];
      const int kb = k0 + j + 16 * half;
      float bv;
      if constexpr (STAGE_B) bv = lB[kb * N + c0 + m];
      else                   bv = Bm[(size_t)kb * N + c0 + m];
      b[j] = (__bf16)bv;
    }
    acc = __builtin_amdgcn_wmma_f32_16x16x32_bf16(
        false, a, false, b, (short)0, acc, false, false);
  }

  const int col = c0 + m;
  const size_t rowBase = (size_t)blockIdx.x * ROWS + rowTile * 16 + 8 * half;
#pragma unroll
  for (int r = 0; r < 8; ++r) {
    float v = acc[r];
    if constexpr (BIAS) v += bias[col];
    if constexpr (ACT == 2) v = fmaxf(v, 0.f);
    D[(rowBase + r) * N + col] = v;
  }
}

// -------------------------- GCN aggregation (F=32) -------------------------
__global__ void gcn_agg_init32_k(const float* __restrict__ h, const float* __restrict__ dinv,
                                 const float* __restrict__ b, float* __restrict__ agg, int n) {
  int i = blockIdx.x * blockDim.x + threadIdx.x;
  if (i >= n * 32) return;
  int node = i >> 5, f = i & 31;
  float di = dinv[node];
  agg[i] = h[i] * di * di + b[f];
}

__global__ void gcn_edge32_k(const int* __restrict__ src, const int* __restrict__ dst,
                             const float* __restrict__ h, const float* __restrict__ dinv,
                             float* __restrict__ agg, int e) {
  int i = blockIdx.x * blockDim.x + threadIdx.x;
  if (i >= e * 32) return;
  int edge = i >> 5, f = i & 31;
  int s = src[edge], d = dst[edge];
  float w = dinv[s] * dinv[d];
  atomicAdd(&agg[(size_t)d * 32 + f], h[(size_t)s * 32 + f] * w);
}

__global__ void tanh_k(const float* __restrict__ in, float* __restrict__ out, int n) {
  int i = blockIdx.x * blockDim.x + threadIdx.x;
  if (i < n) out[i] = tanhf(in[i]);
}

// -------------------------- GCN layer 4 (F=1) ------------------------------
__global__ void matvec32_k(const float* __restrict__ A, const float* __restrict__ w,
                           float* __restrict__ out, int n) {
  int i = blockIdx.x * blockDim.x + threadIdx.x;
  if (i >= n) return;
  float s = 0.f;
#pragma unroll
  for (int j = 0; j < 32; ++j) s += A[(size_t)i * 32 + j] * w[j];
  out[i] = s;
}

__global__ void gcn_agg_init1_k(const float* __restrict__ h, const float* __restrict__ dinv,
                                const float* __restrict__ b, float* __restrict__ agg, int n) {
  int i = blockIdx.x * blockDim.x + threadIdx.x;
  if (i >= n) return;
  float di = dinv[i];
  agg[i] = h[i] * di * di + b[0];
}

__global__ void gcn_edge1_k(const int* __restrict__ src, const int* __restrict__ dst,
                            const float* __restrict__ h, const float* __restrict__ dinv,
                            float* __restrict__ agg, int e) {
  int i = blockIdx.x * blockDim.x + threadIdx.x;
  if (i >= e) return;
  int s = src[i], d = dst[i];
  atomicAdd(&agg[d], h[s] * dinv[s] * dinv[d]);
}

// -------------------------- sort pool --------------------------------------
// One wave per graph. Stable descending top-K selection by h4 (last feature),
// done redundantly per lane; lane-parallel 30x97 feature copy (zero-padded).
__global__ void sortpool_k(const int* __restrict__ batch,
                           const float* __restrict__ h1, const float* __restrict__ h2,
                           const float* __restrict__ h3, const float* __restrict__ h4,
                           float* __restrict__ pooled, int n) {
  const int g = blockIdx.x * (blockDim.x >> 5) + (threadIdx.x >> 5);
  const int lane = threadIdx.x & 31;
  if (g >= NGRAPH) return;

  int lo = 0, hi = n;
  while (lo < hi) { int mid = (lo + hi) >> 1; if (batch[mid] < g) lo = mid + 1; else hi = mid; }
  const int s = lo;
  lo = s; hi = n;
  while (lo < hi) { int mid = (lo + hi) >> 1; if (batch[mid] < g + 1) lo = mid + 1; else hi = mid; }
  const int e = lo;
  const int cnt = e - s;
  const int used = cnt < TOPK ? cnt : TOPK;

  int chosen[TOPK];
  float prevKey = __builtin_inff();
  int prevIdx = -1;
  for (int p = 0; p < used; ++p) {
    float bestKey = -__builtin_inff();
    int bestIdx = -1;
    for (int j = s; j < e; ++j) {
      float kj = h4[j];
      bool after = (kj < prevKey) || (kj == prevKey && j > prevIdx);
      if (after && (kj > bestKey || bestIdx < 0)) { bestKey = kj; bestIdx = j; }
    }
    chosen[p] = bestIdx;
    prevKey = bestKey;
    prevIdx = bestIdx;
  }

  for (int t = lane; t < TOPK * FEAT; t += 32) {
    int p = t / FEAT, f = t % FEAT;
    float v = 0.f;
    if (p < used) {
      int node = chosen[p];
      if (f < 32)      v = h1[(size_t)node * 32 + f];
      else if (f < 64) v = h2[(size_t)node * 32 + f - 32];
      else if (f < 96) v = h3[(size_t)node * 32 + f - 64];
      else             v = h4[node];
    }
    pooled[(size_t)g * (TOPK * FEAT) + t] = v;
  }
}

// -------------------------- CNN / MLP head ---------------------------------
__global__ void conv1_relu_k(const float* __restrict__ pooled, const float* __restrict__ w,
                             const float* __restrict__ b, float* __restrict__ out) {
  int i = blockIdx.x * blockDim.x + threadIdx.x;           // 2048*16*30
  if (i >= NGRAPH * 16 * 30) return;
  int t = i % 30, c = (i / 30) % 16, g = i / (16 * 30);
  float s = b[c];
  const float* pp = pooled + (size_t)g * (TOPK * FEAT) + t * 97;
  const float* ww = w + c * 97;
  for (int k = 0; k < 97; ++k) s += pp[k] * ww[k];
  out[(size_t)g * 480 + c * 30 + t] = fmaxf(s, 0.f);
}

__global__ void maxpool2_k(const float* __restrict__ in, float* __restrict__ out) {
  int i = blockIdx.x * blockDim.x + threadIdx.x;           // 2048*16*15
  if (i >= NGRAPH * 16 * 15) return;
  int t = i % 15, c = (i / 15) % 16, g = i / (16 * 15);
  const float* p = in + (size_t)g * 480 + c * 30 + 2 * t;
  out[(size_t)g * 240 + c * 15 + t] = fmaxf(p[0], p[1]);
}

__global__ void conv2_relu_k(const float* __restrict__ in, const float* __restrict__ w,
                             const float* __restrict__ b, float* __restrict__ out) {
  int i = blockIdx.x * blockDim.x + threadIdx.x;           // 2048*32*11
  if (i >= NGRAPH * 32 * 11) return;
  int t = i % 11, c = (i / 11) % 32, g = i / (32 * 11);
  float s = b[c];
#pragma unroll
  for (int ci = 0; ci < 16; ++ci) {
    const float* p = in + (size_t)g * 240 + ci * 15 + t;
    const float* ww = w + ((size_t)c * 16 + ci) * 5;
#pragma unroll
    for (int k = 0; k < 5; ++k) s += p[k] * ww[k];
  }
  out[(size_t)g * 352 + c * 11 + t] = fmaxf(s, 0.f);
}

__global__ void lin2_sigmoid_k(const float* __restrict__ fc1, const float* __restrict__ w,
                               const float* __restrict__ b, float* __restrict__ out) {
  int g = blockIdx.x * blockDim.x + threadIdx.x;
  if (g >= NGRAPH) return;
  float s = b[0];
#pragma unroll 8
  for (int k = 0; k < 128; ++k) s += fc1[(size_t)g * 128 + k] * w[k];
  out[g] = 1.0f / (1.0f + expf(-s));
}

// ---------------------------------------------------------------------------
extern "C" void kernel_launch(void* const* d_in, const int* in_sizes, int n_in,
                              void* d_out, int out_size, void* d_ws, size_t ws_size,
                              hipStream_t stream) {
  const float* x   = (const float*)d_in[0];
  const int*   eidx= (const int*)d_in[1];
  const int*   batch=(const int*)d_in[2];
  const float* W1  = (const float*)d_in[3];
  const float* b1  = (const float*)d_in[4];
  const float* W2  = (const float*)d_in[5];
  const float* b2  = (const float*)d_in[6];
  const float* W3  = (const float*)d_in[7];
  const float* b3  = (const float*)d_in[8];
  const float* W4  = (const float*)d_in[9];
  const float* b4  = (const float*)d_in[10];
  const float* c1w = (const float*)d_in[11];
  const float* c1b = (const float*)d_in[12];
  const float* c2w = (const float*)d_in[13];
  const float* c2b = (const float*)d_in[14];
  const float* l1w = (const float*)d_in[15];
  const float* l1b = (const float*)d_in[16];
  const float* l2w = (const float*)d_in[17];
  const float* l2b = (const float*)d_in[18];
  float* out = (float*)d_out;

  const int n = in_sizes[0] / 128;   // 102400
  const int e = in_sizes[1] / 2;     // 1638400
  const int* src = eidx;             // edge_index[0]
  const int* dst = eidx + e;         // edge_index[1]

  // workspace layout (floats), ~100 MB total
  float* ws     = (float*)d_ws;
  float* dinv   = ws;                              // n
  float* hpre   = ws + (size_t)n;                  // 32n
  float* agg    = ws + (size_t)33 * n;             // 32n
  float* h1     = ws + (size_t)65 * n;             // 32n
  float* h2     = ws + (size_t)97 * n;             // 32n
  float* h3     = ws + (size_t)129 * n;            // 32n
  float* h4     = ws + (size_t)161 * n;            // n
  float* pooled = ws + (size_t)162 * n;            // 2048*2910
  float* c1o    = pooled + (size_t)NGRAPH * (TOPK * FEAT);
  float* p1o    = c1o + (size_t)NGRAPH * 480;
  float* c2o    = p1o + (size_t)NGRAPH * 240;
  float* fc1    = c2o + (size_t)NGRAPH * 352;

  const int B = 256;

  // deg -> dinv (deg starts at 1.0 for self loop)
  fill_f32_k<<<cdiv(n, B), B, 0, stream>>>(dinv, 1.0f, n);
  deg_accum_k<<<cdiv(e, B), B, 0, stream>>>(dst, dinv, e);
  deg_to_dinv_k<<<cdiv(n, B), B, 0, stream>>>(dinv, n);

  // GCN layer 1: x(n,128) @ W1(128,32)   [64-row stripes, 1600 blocks]
  gemm_bf16_wmma_k<128, 32, 0, false><<<n / 64, B, 0, stream>>>(x, W1, nullptr, hpre);
  gcn_agg_init32_k<<<cdiv(n * 32, B), B, 0, stream>>>(hpre, dinv, b1, agg, n);
  gcn_edge32_k<<<cdiv(e * 32, B), B, 0, stream>>>(src, dst, hpre, dinv, agg, e);
  tanh_k<<<cdiv(n * 32, B), B, 0, stream>>>(agg, h1, n * 32);

  // GCN layer 2
  gemm_bf16_wmma_k<32, 32, 0, false><<<n / 64, B, 0, stream>>>(h1, W2, nullptr, hpre);
  gcn_agg_init32_k<<<cdiv(n * 32, B), B, 0, stream>>>(hpre, dinv, b2, agg, n);
  gcn_edge32_k<<<cdiv(e * 32, B), B, 0, stream>>>(src, dst, hpre, dinv, agg, e);
  tanh_k<<<cdiv(n * 32, B), B, 0, stream>>>(agg, h2, n * 32);

  // GCN layer 3
  gemm_bf16_wmma_k<32, 32, 0, false><<<n / 64, B, 0, stream>>>(h2, W3, nullptr, hpre);
  gcn_agg_init32_k<<<cdiv(n * 32, B), B, 0, stream>>>(hpre, dinv, b3, agg, n);
  gcn_edge32_k<<<cdiv(e * 32, B), B, 0, stream>>>(src, dst, hpre, dinv, agg, e);
  tanh_k<<<cdiv(n * 32, B), B, 0, stream>>>(agg, h3, n * 32);

  // GCN layer 4: (n,32) @ (32,1)
  matvec32_k<<<cdiv(n, B), B, 0, stream>>>(h3, W4, hpre, n);
  gcn_agg_init1_k<<<cdiv(n, B), B, 0, stream>>>(hpre, dinv, b4, agg, n);
  gcn_edge1_k<<<cdiv(e, B), B, 0, stream>>>(src, dst, hpre, dinv, agg, e);
  tanh_k<<<cdiv(n, B), B, 0, stream>>>(agg, h4, n);

  // sort pool: wave per graph
  sortpool_k<<<cdiv(NGRAPH * 32, B), B, 0, stream>>>(batch, h1, h2, h3, h4, pooled, n);

  // head
  conv1_relu_k<<<cdiv(NGRAPH * 16 * 30, B), B, 0, stream>>>(pooled, c1w, c1b, c1o);
  maxpool2_k<<<cdiv(NGRAPH * 16 * 15, B), B, 0, stream>>>(c1o, p1o);
  conv2_relu_k<<<cdiv(NGRAPH * 32 * 11, B), B, 0, stream>>>(p1o, c2w, c2b, c2o);
  // lin1: (2048,352) @ (352,128) + bias, relu   [16-row stripes, 128 blocks]
  gemm_bf16_wmma_k<352, 128, 2, true><<<NGRAPH / 16, B, 0, stream>>>(c2o, l1w, l1b, fc1);
  lin2_sigmoid_k<<<cdiv(NGRAPH, B), B, 0, stream>>>(fc1, l2w, l2b, out);

  (void)n_in; (void)out_size; (void)ws_size;
}
